// TrunkGNN_5360119185623
// MI455X (gfx1250) — matrix operand order repaired and checked by reading, not compiled, not run on hardware
//
#include <hip/hip_runtime.h>

typedef __attribute__((ext_vector_type(16))) _Float16 v16h;
typedef __attribute__((ext_vector_type(8)))  _Float16 v8h;
typedef __attribute__((ext_vector_type(8)))  float    v8f;
typedef __attribute__((ext_vector_type(4)))  unsigned int sgpr4;
typedef __attribute__((ext_vector_type(8)))  unsigned int sgpr8;

static constexpr int       kNodes = 100000;
static constexpr long long kEdges = 1600000;
static constexpr float     kDt    = 0.01f;

// ---- workspace layout ----
// packed f16 weights: 141312 halves = 282624 B  (edge: 117760 halves, node: 23552)
// packed f16 biases : 1104 halves at byte offset 282624
// f32 aggregation   : 100000*50 floats at byte offset 290816
static constexpr size_t WS_BIAS_OFF = 282624;
static constexpr size_t WS_AGGR_OFF = 290816;

// ---------------------------------------------------------------------------
// Tensor Data Mover: DMA `bytes` (multiple of 8, <= 512KB) of contiguous
// global memory into LDS at byte offset lds_off.  D# per CDNA5 ISA ch.8:
//   group0: count=1 | lds_addr | global_addr[56:0] | type=2
//   group1: data_size=8B, tensor_dim0=tile_dim0=n8, tensor_dim1=tile_dim1=1,
//           tensor_dim0_stride=n8, no pad/iterate/multicast.
// Issued by one wave; completion via s_wait_tensorcnt.
// ---------------------------------------------------------------------------
__device__ __forceinline__ void tdm_load_to_lds(const void* gsrc, unsigned lds_off,
                                                unsigned bytes) {
  unsigned long long ga = (unsigned long long)(uintptr_t)gsrc;
  unsigned n8 = bytes >> 3;
  sgpr4 g0;
  g0[0] = 1u;                                            // count=1, user mode
  g0[1] = lds_off + (unsigned)__builtin_amdgcn_groupstaticsize();
  g0[2] = (unsigned)ga;                                  // global_addr[31:0]
  g0[3] = ((unsigned)(ga >> 32) & 0x1FFFFFFu) | (2u << 30); // addr[56:32]|type=2
  sgpr8 g1;
  g1[0] = 3u << 16;                                      // data_size = 8 bytes
  g1[1] = (n8 & 0xFFFFu) << 16;                          // tensor_dim0[15:0]
  g1[2] = (n8 >> 16) | (1u << 16);                       // dim0[31:16] | tensor_dim1=1
  g1[3] = n8 << 16;                                      // tile_dim0
  g1[4] = 1u;                                            // tile_dim1=1, tile_dim2=0
  g1[5] = n8;                                            // tensor_dim0_stride[31:0]
  g1[6] = 0u;
  g1[7] = 0u;
  asm volatile("tensor_load_to_lds %0, %1" :: "s"(g0), "s"(g1) : "memory");
}

// ---------------------------------------------------------------------------
// Pack fp32 weights into WMMA B-fragment layout (f16), zero-padded.
// Fragment block (kt,nt) = 1024 B: per lane L (0..31), 16 halves h:
//   K = kt*32 + (h>=8 ? 16 : 0) + (L>=16 ? 8 : 0) + (h&7) ,  N = nt*16 + (L&15)
// ---------------------------------------------------------------------------
__global__ __launch_bounds__(256)
void pack_layer_kernel(const float* __restrict__ W, const float* __restrict__ B,
                       _Float16* __restrict__ dW, _Float16* __restrict__ dB,
                       int fanin, int fanout, int ktiles, int ntiles) {
  int total = ktiles * ntiles * 512;
  int nb = ntiles * 16;
  int stride = gridDim.x * blockDim.x;
  for (int idx = blockIdx.x * blockDim.x + threadIdx.x; idx < total + nb; idx += stride) {
    if (idx < total) {
      int blk = idx >> 9, w = idx & 511;
      int lane = w >> 4, h = w & 15;
      int kt = blk / ntiles, nt = blk - kt * ntiles;
      int K = kt * 32 + ((h >= 8) ? 16 : 0) + ((lane >= 16) ? 8 : 0) + (h & 7);
      int N = nt * 16 + (lane & 15);
      float v = (K < fanin && N < fanout) ? W[K * fanout + N] : 0.f;
      dW[idx] = (_Float16)v;
    } else {
      int n = idx - total;
      dB[n] = (_Float16)((n < fanout) ? B[n] : 0.f);
    }
  }
}

__global__ __launch_bounds__(256)
void zero_kernel(float* __restrict__ p, int n) {
  int stride = gridDim.x * blockDim.x;
  for (int i = blockIdx.x * blockDim.x + threadIdx.x; i < n; i += stride) p[i] = 0.f;
}

// ---------------------------------------------------------------------------
// Edge kernel: 128 edges/block (8 waves x 16-edge M-tiles).
// LDS: actA/actB 128x160 f16 (40960 B each), wb0/wb1 25600 halves each
// (TDM double buffer), recvs[128].  Total 184832 B.
// ---------------------------------------------------------------------------
__global__ __launch_bounds__(256)
void edge_kernel(const float* __restrict__ x,
                 const long long* __restrict__ eidx,
                 const _Float16* __restrict__ wpack,
                 const _Float16* __restrict__ bias,
                 float* __restrict__ aggr) {
  extern __shared__ __attribute__((aligned(32))) char smem[];
  _Float16* actA = (_Float16*)smem;                 // 128*160 halves
  _Float16* actB = actA + 128 * 160;
  _Float16* wb0  = actB + 128 * 160;                // byte offset 81920
  _Float16* wb1  = wb0 + 25600;                     // byte offset 133120
  int* recvs = (int*)(wb1 + 25600);                 // byte offset 184320

  const long long* sender   = eidx;
  const long long* receiver = eidx + kEdges;

  const int tid  = threadIdx.x;
  const int lane = tid & 31;
  const int wave = tid >> 5;
  const long long ebase = (long long)blockIdx.x * 128;

  // kick off DMA of layer-0 weights while we stage activations
  if (wave == 0) tdm_load_to_lds(wpack, 81920u, 10240u);

  // stage diff = x[recv]-x[send], padded K 6->32
  for (int i = tid; i < 128 * 32; i += 256) {
    int e = i >> 5, k = i & 31;
    long long eg = ebase + e;
    float v = 0.f;
    if (eg < kEdges && k < 6) {
      long long s = sender[eg];
      long long r = receiver[eg];
      v = x[r * 6 + k] - x[s * 6 + k];
    }
    actA[i] = (_Float16)v;
  }
  if (tid < 128) {
    long long eg = ebase + tid;
    recvs[tid] = (eg < kEdges) ? (int)receiver[eg] : -1;
  }

  const int mloc = lane & 15;
  const int koff = (lane < 16) ? 0 : 8;
  const int m0   = (wave << 4) + ((lane < 16) ? 0 : 8);

  _Float16* cur = actA;
  _Float16* nxt = actB;

  // all DMAs issued so far complete + activations/epilogues published
  auto sync_wait = [&]() {
    if (wave == 0) __builtin_amdgcn_s_wait_tensorcnt(0);
    __syncthreads();
  };

  // one linear layer from weight buffer wb (no barriers inside)
  auto run = [&](const _Float16* wb, const _Float16* bsrc,
                 int kp, int ktiles, int ntiles, bool hidden) {
    const _Float16* arow = cur + ((wave << 4) + mloc) * kp;
    const int np = ntiles << 4;
    for (int nt = 0; nt < ntiles; ++nt) {
      v8f acc = {0.f, 0.f, 0.f, 0.f, 0.f, 0.f, 0.f, 0.f};
      for (int kt = 0; kt < ktiles; ++kt) {
        union { v16h v; v8h h[2]; } ua;
        ua.h[0] = *(const v8h*)(arow + kt * 32 + koff);
        ua.h[1] = *(const v8h*)(arow + kt * 32 + 16 + koff);
        v16h bf = *(const v16h*)(wb + ((kt * ntiles + nt) << 9) + (lane << 4));
        acc = __builtin_amdgcn_wmma_f32_16x16x32_f16(false, ua.v, false, bf,
                                                     (short)0, acc, false, false);
      }
      const int n  = (nt << 4) + mloc;
      const float bv = (float)bsrc[n];
      if (hidden) {
        #pragma unroll
        for (int r = 0; r < 8; ++r) {
          float v = acc[r] + bv;
          nxt[(m0 + r) * np + n] = (_Float16)(v > 0.f ? v : 0.f);
        }
      } else if (n < 50) {
        #pragma unroll
        for (int r = 0; r < 8; ++r) {
          int rc = recvs[m0 + r];
          if (rc >= 0) atomicAdd(aggr + (long long)rc * 50 + n, acc[r] + bv);
        }
      }
    }
    _Float16* t = cur; cur = nxt; nxt = t;
  };

  sync_wait();                                          // wb0 ready, actA staged

  // L0 (6->150) from wb0; prefetch L1 into wb1 underneath
  if (wave == 0) tdm_load_to_lds(wpack + 5120, 133120u, 51200u);
  run(wb0, bias, 32, 1, 10, true);
  sync_wait();

  // L1..L4 (150->150); layer l reads wb[l&1], prefetch l+1 into wb[(l+1)&1]
  for (int l = 1; l <= 4; ++l) {
    const _Float16* wnext = (l < 4) ? (wpack + 5120 + l * 25600) : (wpack + 107520);
    unsigned nextOff   = ((l + 1) & 1) ? 133120u : 81920u;
    unsigned nextBytes = (l < 4) ? 51200u : 20480u;
    if (wave == 0) tdm_load_to_lds(wnext, nextOff, nextBytes);
    run((l & 1) ? wb1 : wb0, bias + l * 160, 160, 5, 10, true);
    sync_wait();
  }

  // L5 (150->50) from wb1, scatter-add straight from accumulators
  run(wb1, bias + 800, 160, 5, 4, false);
}

// ---------------------------------------------------------------------------
// Node kernel: 128 nodes/block. [x||aggr] (56->pad 64) -> 100 -> 100 -> 3,
// then symplectic Euler update into d_out.
// LDS: act0/act1 128x128 halves (32768 B each), wb0/wb1 14336 halves each.
// Total 122880 B.
// ---------------------------------------------------------------------------
__global__ __launch_bounds__(256)
void node_kernel(const float* __restrict__ x,
                 const _Float16* __restrict__ wpack,
                 const _Float16* __restrict__ bias,
                 const float* __restrict__ aggr,
                 float* __restrict__ out) {
  extern __shared__ __attribute__((aligned(32))) char smem[];
  _Float16* act0 = (_Float16*)smem;       // 128*128 halves
  _Float16* act1 = act0 + 128 * 128;
  _Float16* wb0  = act1 + 128 * 128;      // byte offset 65536
  _Float16* wb1  = wb0 + 14336;           // byte offset 94208

  const int tid  = threadIdx.x;
  const int lane = tid & 31;
  const int wave = tid >> 5;
  const long long nbase = (long long)blockIdx.x * 128;

  if (wave == 0) tdm_load_to_lds(wpack + 117760, 65536u, 14336u);  // L0 weights

  // zero pad regions so K-padding reads are clean
  for (int i = tid; i < 128 * 128; i += 256) act1[i] = (_Float16)0.f;
  for (int i = 8192 + tid; i < 128 * 128; i += 256) act0[i] = (_Float16)0.f;
  // stage [x || aggr], padded K 56->64 (stride 64 in act0's first 8192 halves)
  for (int i = tid; i < 128 * 64; i += 256) {
    int nd = i >> 6, k = i & 63;
    long long g = nbase + nd;
    float v = 0.f;
    if (g < kNodes && k < 56) v = (k < 6) ? x[g * 6 + k] : aggr[g * 50 + (k - 6)];
    act0[i] = (_Float16)v;
  }

  const int mloc = lane & 15;
  const int koff = (lane < 16) ? 0 : 8;
  const int m0   = (wave << 4) + ((lane < 16) ? 0 : 8);

  auto sync_wait = [&]() {
    if (wave == 0) __builtin_amdgcn_s_wait_tensorcnt(0);
    __syncthreads();
  };

  auto hidden_layer = [&](const _Float16* wb, const _Float16* bsrc,
                          const _Float16* src, _Float16* dst,
                          int kp, int ktiles, int ntiles) {
    const _Float16* arow = src + ((wave << 4) + mloc) * kp;
    for (int nt = 0; nt < ntiles; ++nt) {
      v8f acc = {0.f, 0.f, 0.f, 0.f, 0.f, 0.f, 0.f, 0.f};
      for (int kt = 0; kt < ktiles; ++kt) {
        union { v16h v; v8h h[2]; } ua;
        ua.h[0] = *(const v8h*)(arow + kt * 32 + koff);
        ua.h[1] = *(const v8h*)(arow + kt * 32 + 16 + koff);
        v16h bf = *(const v16h*)(wb + ((kt * ntiles + nt) << 9) + (lane << 4));
        acc = __builtin_amdgcn_wmma_f32_16x16x32_f16(false, ua.v, false, bf,
                                                     (short)0, acc, false, false);
      }
      const int n  = (nt << 4) + mloc;
      const float bv = (float)bsrc[n];
      #pragma unroll
      for (int r = 0; r < 8; ++r) {
        float v = acc[r] + bv;
        dst[(m0 + r) * 128 + n] = (_Float16)(v > 0.f ? v : 0.f);  // stride 128
      }
    }
  };

  sync_wait();                                             // wb0 ready, input staged

  if (wave == 0) tdm_load_to_lds(wpack + 124928, 94208u, 28672u);  // L1 -> wb1
  hidden_layer(wb0, bias + 864, act0, act1, 64, 2, 7);     // 56->100
  sync_wait();

  if (wave == 0) tdm_load_to_lds(wpack + 139264, 65536u, 4096u);   // L2 -> wb0
  hidden_layer(wb1, bias + 976, act1, act0, 128, 4, 7);    // 100->100
  sync_wait();

  // final layer: 100->3 (kp=128, ktiles=4, ntiles=1) + integration
  {
    const _Float16* arow = act0 + ((wave << 4) + mloc) * 128;
    v8f acc = {0.f, 0.f, 0.f, 0.f, 0.f, 0.f, 0.f, 0.f};
    for (int kt = 0; kt < 4; ++kt) {
      union { v16h v; v8h h[2]; } ua;
      ua.h[0] = *(const v8h*)(arow + kt * 32 + koff);
      ua.h[1] = *(const v8h*)(arow + kt * 32 + 16 + koff);
      v16h bf = *(const v16h*)(wb0 + (kt << 9) + (lane << 4));
      acc = __builtin_amdgcn_wmma_f32_16x16x32_f16(false, ua.v, false, bf,
                                                   (short)0, acc, false, false);
    }
    const int n = mloc;
    if (n < 3) {
      const float bv = (float)bias[1088 + n];
      #pragma unroll
      for (int r = 0; r < 8; ++r) {
        long long g = nbase + m0 + r;
        if (g < kNodes) {
          float dv   = acc[r] + bv;
          float vnew = x[g * 6 + 3 + n] + dv;
          out[g * 6 + n]     = x[g * 6 + n] + vnew * kDt;
          out[g * 6 + 3 + n] = vnew;
        }
      }
    }
  }
}

// ---------------------------------------------------------------------------
extern "C" void kernel_launch(void* const* d_in, const int* in_sizes, int n_in,
                              void* d_out, int out_size, void* d_ws, size_t ws_size,
                              hipStream_t stream) {
  (void)in_sizes; (void)n_in; (void)out_size; (void)ws_size;
  const float*     x    = (const float*)d_in[0];
  const long long* eidx = (const long long*)d_in[1];   // int64 [2, E]
  _Float16* wp   = (_Float16*)d_ws;
  _Float16* bp   = (_Float16*)((char*)d_ws + WS_BIAS_OFF);
  float*    aggr = (float*)((char*)d_ws + WS_AGGR_OFF);
  float*    out  = (float*)d_out;

  struct L { int in_idx, fanin, fanout, ktiles, ntiles, woff, boff; };
  const L layers[9] = {
    { 2,   6, 150, 1, 10,      0,    0},
    { 4, 150, 150, 5, 10,   5120,  160},
    { 6, 150, 150, 5, 10,  30720,  320},
    { 8, 150, 150, 5, 10,  56320,  480},
    {10, 150, 150, 5, 10,  81920,  640},
    {12, 150,  50, 5,  4, 107520,  800},
    {14,  56, 100, 2,  7, 117760,  864},
    {16, 100, 100, 4,  7, 124928,  976},
    {18, 100,   3, 4,  1, 139264, 1088},
  };
  for (int i = 0; i < 9; ++i) {
    const L& l = layers[i];
    pack_layer_kernel<<<dim3(64), dim3(256), 0, stream>>>(
        (const float*)d_in[l.in_idx], (const float*)d_in[l.in_idx + 1],
        wp + l.woff, bp + l.boff, l.fanin, l.fanout, l.ktiles, l.ntiles);
  }
  zero_kernel<<<dim3(512), dim3(256), 0, stream>>>(aggr, kNodes * 50);

  edge_kernel<<<dim3((int)((kEdges + 127) / 128)), dim3(256), 184832, stream>>>(
      x, eidx, wp, bp, aggr);
  node_kernel<<<dim3((kNodes + 127) / 128), dim3(256), 122880, stream>>>(
      x, wp, bp, aggr, out);
}
